// INN_MultiCodebook_44976897523909
// MI455X (gfx1250) — matrix operand order repaired and checked by reading, not compiled, not run on hardware
//
#include <hip/hip_runtime.h>
#include <cstdint>
#include <cstddef>

// ---------------------------------------------------------------------------
// Types for CDNA5 WMMA (wave32): bf16 16x16x32, f32 accumulate
// ---------------------------------------------------------------------------
typedef __attribute__((ext_vector_type(16))) __bf16   v16bf;
typedef __attribute__((ext_vector_type(8)))  float    v8f;
typedef __attribute__((ext_vector_type(4)))  unsigned int u32x4;

union Frag16 {
    v16bf v;
    u32x4 q[2];
};

// Problem constants
#define BB 8
#define KK 4
#define TT 1024
#define VV 2048
#define DD 512
#define NN 8
#define HH 512
#define NHD 8
#define HD 64
#define G4H 2048   // 4*H

// ---------------------------------------------------------------------------
// Generic helpers
// ---------------------------------------------------------------------------
__device__ inline float block_reduce_sum(float v, float* red, int tid, int nthreads) {
    red[tid] = v;
    __syncthreads();
    for (int s = nthreads >> 1; s > 0; s >>= 1) {
        if (tid < s) red[tid] += red[tid + s];
        __syncthreads();
    }
    float r = red[0];
    __syncthreads();
    return r;
}

__device__ inline float sigmoidf_(float x) { return 1.0f / (1.0f + expf(-x)); }

// ---------------------------------------------------------------------------
// f32 -> bf16 flat convert
// ---------------------------------------------------------------------------
__global__ __launch_bounds__(256) void cvt_f32_bf16_kernel(const float* __restrict__ src,
                                                           __bf16* __restrict__ dst, int n) {
    int i = blockIdx.x * 256 + threadIdx.x;
    if (i < n) dst[i] = (__bf16)src[i];
}

// Pack w_ih|w_hh into wcat[n][2048][1024] bf16 (K = D || H)
__global__ __launch_bounds__(256) void wcat_kernel(const float* __restrict__ wih,
                                                   const float* __restrict__ whh,
                                                   __bf16* __restrict__ wcat) {
    int i = blockIdx.x * 256 + threadIdx.x;            // 0 .. 8*2048*1024-1
    if (i >= NN * G4H * 1024) return;
    int row = i >> 10;                                 // n*2048+g
    int j   = i & 1023;
    float v = (j < 512) ? wih[(size_t)row * 512 + j] : whh[(size_t)row * 512 + (j - 512)];
    wcat[i] = (__bf16)v;
}

// ---------------------------------------------------------------------------
// Embedding sum + LayerNorm.  One block per (b,t); writes f32 [b*T+t][D],
// bf16 [b*T+t][D], and scan-ordered bf16 [t*8+b][D].
// ---------------------------------------------------------------------------
__global__ __launch_bounds__(256) void embed_ln_kernel(const int* __restrict__ x,
                                                       const float* __restrict__ emb,
                                                       const float* __restrict__ g,
                                                       const float* __restrict__ bvec,
                                                       float* __restrict__ xf_f32,
                                                       __bf16* __restrict__ xf_bt,
                                                       __bf16* __restrict__ xf_tb) {
    __shared__ float red[256];
    __shared__ int idxs[KK];
    int r = blockIdx.x;                 // b*T + t
    int b = r >> 10, t = r & 1023;
    int tid = threadIdx.x;
    if (tid < KK) idxs[tid] = x[((size_t)b * KK + tid) * TT + t];
    __syncthreads();
    int d0 = tid * 2;
    float v0 = 0.f, v1 = 0.f;
#pragma unroll
    for (int k = 0; k < KK; ++k) {
        const float* e = emb + ((size_t)k * (VV + 1) + idxs[k]) * DD;
        v0 += e[d0];
        v1 += e[d0 + 1];
    }
    float s = block_reduce_sum(v0 + v1, red, tid, 256);
    float mean = s * (1.0f / DD);
    float c0 = v0 - mean, c1 = v1 - mean;
    float vs = block_reduce_sum(c0 * c0 + c1 * c1, red, tid, 256);
    float inv = rsqrtf(vs * (1.0f / DD) + 1e-5f);
    float y0 = c0 * inv * g[d0] + bvec[d0];
    float y1 = c1 * inv * g[d0 + 1] + bvec[d0 + 1];
    size_t o = (size_t)r * DD + d0;
    xf_f32[o] = y0; xf_f32[o + 1] = y1;
    xf_bt[o] = (__bf16)y0; xf_bt[o + 1] = (__bf16)y1;
    size_t o2 = ((size_t)(t * BB + b)) * DD + d0;
    xf_tb[o2] = (__bf16)y0; xf_tb[o2 + 1] = (__bf16)y1;
}

// ---------------------------------------------------------------------------
// Generic bf16 WMMA GEMM:  C[M,N] = A[M,K] * W[N,K]^T + bias
// One wave (32 threads) per block; each wave computes a 16x64 tile of C.
// A fragment layout per CDNA5 ISA: lanes 0-15 row M=lane, K {0-7,16-23};
// lanes 16-31 row M=lane-16, K {8-15,24-31}.  Each group is 16 B contiguous.
// mode: 0 = f32 out at C[row*ldc+col]; 1 = bf16 out; 2 = head scatter
//       out[((b*4+k)*1024+t)*2048+v] with row=b*1024+t, col=k*2048+v.
// ---------------------------------------------------------------------------
__global__ __launch_bounds__(32) void gemm_bf16_wmma_kernel(const __bf16* __restrict__ A, int lda,
                                                            const __bf16* __restrict__ W, int ldb,
                                                            const float* __restrict__ bias,
                                                            void* __restrict__ C, int ldc,
                                                            int Kd, int mode) {
    int lane = threadIdx.x;
    int hf = lane >> 4;          // K-half select
    int rr = lane & 15;          // row (A) / col (B,C) within tile
    int mbase = blockIdx.y * 16;
    int nbase = blockIdx.x * 64;

    v8f z = {0.f, 0.f, 0.f, 0.f, 0.f, 0.f, 0.f, 0.f};
    v8f acc[4];
#pragma unroll
    for (int j = 0; j < 4; ++j) acc[j] = z;

    const __bf16* arow = A + (size_t)(mbase + rr) * lda;
    int kiters = Kd >> 5;
    for (int ks = 0; ks < kiters; ++ks) {
        int k0 = (ks << 5) + (hf << 3);
        Frag16 a;
        a.q[0] = *(const u32x4*)(arow + k0);
        a.q[1] = *(const u32x4*)(arow + k0 + 16);
        int kb = (ks << 5) + (hf << 3);
#pragma unroll
        for (int j = 0; j < 4; ++j) {
            const __bf16* wrow = W + (size_t)(nbase + j * 16 + rr) * ldb + kb;
            Frag16 bm;
            bm.q[0] = *(const u32x4*)(wrow);
            bm.q[1] = *(const u32x4*)(wrow + 16);
            acc[j] = __builtin_amdgcn_wmma_f32_16x16x32_bf16(
                false, a.v, false, bm.v, (short)0, acc[j], false, false);
        }
    }

#pragma unroll
    for (int j = 0; j < 4; ++j) {
        int col = nbase + j * 16 + rr;
        float bv = bias ? bias[col] : 0.0f;
#pragma unroll
        for (int e = 0; e < 8; ++e) {
            int row = mbase + e + hf * 8;     // C layout: lanes16-31 hold M=8+e
            float val = acc[j][e] + bv;
            if (mode == 0) {
                ((float*)C)[(size_t)row * ldc + col] = val;
            } else if (mode == 1) {
                ((__bf16*)C)[(size_t)row * ldc + col] = (__bf16)val;
            } else {
                int b = row >> 10, t = row & 1023;
                int k = col >> 11, v = col & 2047;
                ((float*)C)[(((size_t)(b * KK + k)) * TT + t) * VV + v] = val;
            }
        }
    }
}

// ---------------------------------------------------------------------------
// LSTM scan: one workgroup (512 threads = 16 waves) per neuron n.
// Per step: gates[8,2048] = [x_t | h] (8x1024) @ wcat[n]^T via WMMA
// (M padded to 16, batch in M), then elementwise LSTM update in LDS.
// x_t streamed into LDS via CDNA5 async global->LDS copy.
// ---------------------------------------------------------------------------
__global__ __launch_bounds__(512) void lstm_scan_kernel(const __bf16* __restrict__ xf_tb,
                                                        const __bf16* __restrict__ wcat,
                                                        const float* __restrict__ b_ih,
                                                        const float* __restrict__ b_hh,
                                                        __bf16* __restrict__ hs) {
    // LDS layout (single block so async LDS byte offsets are well defined):
    //   [0,16384)        a_tile : bf16 [8][1024]  ([x | h] rows b)
    //   [16384,49152)    gates  : bf16 [8][2048]
    //   [49152,57344)    biasl  : f32  [2048]
    __shared__ __align__(16) char smem[16384 + 32768 + 8192];
    __bf16* a_tile = (__bf16*)smem;
    __bf16* gates  = (__bf16*)(smem + 16384);
    float*  biasl  = (float*)(smem + 16384 + 32768);

    int n = blockIdx.x;
    int tid = threadIdx.x;
    int wave = tid >> 5, lane = tid & 31;
    int hf = lane >> 4, rr = lane & 15;

    for (int i = tid; i < G4H; i += 512)
        biasl[i] = b_ih[n * G4H + i] + b_hh[n * G4H + i];
    for (int i = tid; i < BB * HH; i += 512)
        a_tile[(i >> 9) * 1024 + 512 + (i & 511)] = (__bf16)0.0f;

    float c[8];
#pragma unroll
    for (int q = 0; q < 8; ++q) c[q] = 0.0f;

    const __bf16* wn = wcat + (size_t)n * G4H * 1024;
    v8f z = {0.f, 0.f, 0.f, 0.f, 0.f, 0.f, 0.f, 0.f};
    __syncthreads();

    for (int t = 0; t < TT; ++t) {
        // ---- stream x_t (8x512 bf16 = 8 KB) into a_tile[:, 0:512] (async) ----
        {
            uint32_t ldsoff = (uint32_t)(((tid >> 6) * 1024 + (tid & 63) * 8) * 2);
            const __bf16* gp = xf_tb + (size_t)t * (BB * DD) + tid * 8;
            unsigned long long ga = (unsigned long long)(uintptr_t)gp;
            asm volatile("global_load_async_to_lds_b128 %0, %1, off"
                         :: "v"(ldsoff), "v"(ga) : "memory");
            asm volatile("s_wait_asynccnt 0x0" ::: "memory");
        }
        __syncthreads();

        // ---- WMMA: each wave owns 8 of the 128 gate column tiles ----
        v8f acc[8];
#pragma unroll
        for (int j = 0; j < 8; ++j) acc[j] = z;

        for (int ks = 0; ks < 32; ++ks) {           // K = 1024
            Frag16 a;
            int k0 = (ks << 5) + (hf << 3);
            if (rr < 8) {                            // rows 8..15 of padded M are zero
                a.q[0] = *(const u32x4*)(a_tile + rr * 1024 + k0);
                a.q[1] = *(const u32x4*)(a_tile + rr * 1024 + k0 + 16);
            } else {
                a.q[0] = (u32x4){0u, 0u, 0u, 0u};
                a.q[1] = (u32x4){0u, 0u, 0u, 0u};
            }
            int kb = (ks << 5) + (hf << 3);
#pragma unroll
            for (int j = 0; j < 8; ++j) {
                int col = (wave * 8 + j) * 16 + rr;
                const __bf16* wr = wn + (size_t)col * 1024 + kb;
                Frag16 bm;
                bm.q[0] = *(const u32x4*)(wr);
                bm.q[1] = *(const u32x4*)(wr + 16);
                acc[j] = __builtin_amdgcn_wmma_f32_16x16x32_bf16(
                    false, a.v, false, bm.v, (short)0, acc[j], false, false);
            }
        }
        // C layout: lanes 0-15, VGPR e -> (M=e, N=lane); rows 0..7 are the batch
        if (hf == 0) {
#pragma unroll
            for (int j = 0; j < 8; ++j) {
                int col = (wave * 8 + j) * 16 + rr;
#pragma unroll
                for (int e = 0; e < 8; ++e)
                    gates[e * G4H + col] = (__bf16)acc[j][e];
            }
        }
        __syncthreads();

        // ---- elementwise LSTM update (gate order i,f,g,o) ----
#pragma unroll
        for (int q = 0; q < 8; ++q) {
            int e = tid + (q << 9);          // 0..4095, unique per (tid,q)
            int b = e >> 9, j = e & 511;
            const __bf16* gb = gates + b * G4H;
            float gi = (float)gb[j]         + biasl[j];
            float gf = (float)gb[512 + j]   + biasl[512 + j];
            float gg = (float)gb[1024 + j]  + biasl[1024 + j];
            float go = (float)gb[1536 + j]  + biasl[1536 + j];
            c[q] = sigmoidf_(gf) * c[q] + sigmoidf_(gi) * tanhf(gg);
            float h = sigmoidf_(go) * tanhf(c[q]);
            a_tile[b * 1024 + 512 + j] = (__bf16)h;
            hs[(((size_t)t * BB + b) * NN + n) * HH + j] = (__bf16)h;
        }
        __syncthreads();
    }
}

// ---------------------------------------------------------------------------
// Per-neuron LayerNorm of proj -> stack (bf16), with layout permutation:
// proj_ws[(t*8+b)][n*512+d]  ->  stack[((b*T+t)*8+n)][d]
// ---------------------------------------------------------------------------
__global__ __launch_bounds__(128) void stack_ln_kernel(const float* __restrict__ proj,
                                                       const float* __restrict__ g,
                                                       const float* __restrict__ bvec,
                                                       __bf16* __restrict__ stack) {
    __shared__ float red[128];
    int bid = blockIdx.x;                 // 65536
    int n = bid & 7;
    int rrow = bid >> 3;                  // t*8+b
    int t = rrow >> 3, b = rrow & 7;
    int tid = threadIdx.x;
    const float* row = proj + (size_t)rrow * (NN * DD) + n * DD;
    float v[4], s = 0.f;
#pragma unroll
    for (int i = 0; i < 4; ++i) { v[i] = row[tid * 4 + i]; s += v[i]; }
    float mean = block_reduce_sum(s, red, tid, 128) * (1.0f / DD);
    float vs = 0.f;
#pragma unroll
    for (int i = 0; i < 4; ++i) { v[i] -= mean; vs += v[i] * v[i]; }
    float inv = rsqrtf(block_reduce_sum(vs, red, tid, 128) * (1.0f / DD) + 1e-5f);
    __bf16* out = stack + (((size_t)(b * TT + t)) * NN + n) * DD;
    const float* gn = g + n * DD;
    const float* bn = bvec + n * DD;
#pragma unroll
    for (int i = 0; i < 4; ++i) {
        int d = tid * 4 + i;
        out[d] = (__bf16)(v[i] * inv * gn[d] + bn[d]);
    }
}

// ---------------------------------------------------------------------------
// Tiny cross-neuron attention: query_len=1, kv_len=N=8. One block per (b,t).
// ---------------------------------------------------------------------------
__global__ __launch_bounds__(256) void attn_kernel(const float* __restrict__ q_ws,
                                                   const __bf16* __restrict__ kk,
                                                   const __bf16* __restrict__ vv,
                                                   __bf16* __restrict__ ao) {
    __shared__ float qs[DD];
    __shared__ float ks[NN * DD];
    __shared__ float vs[NN * DD];
    __shared__ float sc[NHD * NN];
    __shared__ float pr[NHD * NN];
    int r = blockIdx.x;                   // b*T + t
    int tid = threadIdx.x;
    qs[tid] = q_ws[(size_t)r * DD + tid];
    qs[tid + 256] = q_ws[(size_t)r * DD + tid + 256];
    for (int i = tid; i < NN * DD; i += 256) {
        ks[i] = (float)kk[(size_t)r * NN * DD + i];
        vs[i] = (float)vv[(size_t)r * NN * DD + i];
    }
    __syncthreads();
    if (tid < NHD * NN) {
        int h = tid >> 3, n = tid & 7;
        float s = 0.f;
#pragma unroll
        for (int d = 0; d < HD; ++d) s += qs[h * HD + d] * ks[n * DD + h * HD + d];
        sc[tid] = s * 0.125f;             // 1/sqrt(64)
    }
    __syncthreads();
    if (tid < NHD) {
        int h = tid;
        float m = -1e30f;
        for (int n = 0; n < NN; ++n) m = fmaxf(m, sc[h * NN + n]);
        float ssum = 0.f;
        for (int n = 0; n < NN; ++n) { float e = expf(sc[h * NN + n] - m); pr[h * NN + n] = e; ssum += e; }
        float invs = 1.0f / ssum;
        for (int n = 0; n < NN; ++n) pr[h * NN + n] *= invs;
    }
    __syncthreads();
    for (int e = tid; e < DD; e += 256) {
        int h = e >> 6;
        float s = 0.f;
#pragma unroll
        for (int n = 0; n < NN; ++n) s += pr[h * NN + n] * vs[n * DD + e];
        ao[(size_t)r * DD + e] = (__bf16)s;
    }
}

// ---------------------------------------------------------------------------
// mixed = LN(ao@Wo + x_fused) -> bf16
// ---------------------------------------------------------------------------
__global__ __launch_bounds__(256) void mixed_ln_kernel(const float* __restrict__ aoo,
                                                       const float* __restrict__ xf_f32,
                                                       const float* __restrict__ g,
                                                       const float* __restrict__ bvec,
                                                       __bf16* __restrict__ mixed) {
    __shared__ float red[256];
    int r = blockIdx.x;
    int tid = threadIdx.x;
    int d0 = tid * 2;
    size_t o = (size_t)r * DD + d0;
    float v0 = aoo[o] + xf_f32[o];
    float v1 = aoo[o + 1] + xf_f32[o + 1];
    float mean = block_reduce_sum(v0 + v1, red, tid, 256) * (1.0f / DD);
    float c0 = v0 - mean, c1 = v1 - mean;
    float inv = rsqrtf(block_reduce_sum(c0 * c0 + c1 * c1, red, tid, 256) * (1.0f / DD) + 1e-5f);
    mixed[o]     = (__bf16)(c0 * inv * g[d0] + bvec[d0]);
    mixed[o + 1] = (__bf16)(c1 * inv * g[d0 + 1] + bvec[d0 + 1]);
}

// ---------------------------------------------------------------------------
// Launcher
// ---------------------------------------------------------------------------
extern "C" void kernel_launch(void* const* d_in, const int* in_sizes, int n_in,
                              void* d_out, int out_size, void* d_ws, size_t ws_size,
                              hipStream_t stream) {
    (void)in_sizes; (void)n_in; (void)out_size; (void)ws_size;
    const int*   x       = (const int*)  d_in[0];
    const float* emb     = (const float*)d_in[1];
    const float* ln_in_g = (const float*)d_in[2];
    const float* ln_in_b = (const float*)d_in[3];
    const float* w_ih    = (const float*)d_in[4];
    const float* w_hh    = (const float*)d_in[5];
    const float* b_ih    = (const float*)d_in[6];
    const float* b_hh    = (const float*)d_in[7];
    const float* w_proj  = (const float*)d_in[8];
    const float* b_proj  = (const float*)d_in[9];
    const float* ln_n_g  = (const float*)d_in[10];
    const float* ln_n_b  = (const float*)d_in[11];
    const float* wq = (const float*)d_in[12]; const float* bq = (const float*)d_in[13];
    const float* wk = (const float*)d_in[14]; const float* bk = (const float*)d_in[15];
    const float* wv = (const float*)d_in[16]; const float* bv = (const float*)d_in[17];
    const float* wo = (const float*)d_in[18]; const float* bo = (const float*)d_in[19];
    const float* ln_mix_g = (const float*)d_in[20];
    const float* ln_mix_b = (const float*)d_in[21];
    const float* w_head = (const float*)d_in[22];
    const float* b_head = (const float*)d_in[23];

    char* ws = (char*)d_ws;
    size_t off = 0;
    auto wsalloc = [&](size_t bytes) -> char* {
        char* p = ws + off;
        off += (bytes + 255) & ~(size_t)255;
        return p;
    };
    float*  xf_f32  = (float*) wsalloc((size_t)8192 * 512 * 4);     // [b*T+t][D] f32
    __bf16* xf_bt   = (__bf16*)wsalloc((size_t)8192 * 512 * 2);     // [b*T+t][D] bf16
    __bf16* xf_tb   = (__bf16*)wsalloc((size_t)8192 * 512 * 2);     // [t*8+b][D] bf16
    __bf16* wcat    = (__bf16*)wsalloc((size_t)8 * 2048 * 1024 * 2);
    __bf16* wprojb  = (__bf16*)wsalloc((size_t)8 * 512 * 512 * 2);
    __bf16* wqb     = (__bf16*)wsalloc((size_t)512 * 512 * 2);
    __bf16* wkb     = (__bf16*)wsalloc((size_t)512 * 512 * 2);
    __bf16* wvb     = (__bf16*)wsalloc((size_t)512 * 512 * 2);
    __bf16* wob     = (__bf16*)wsalloc((size_t)512 * 512 * 2);
    __bf16* wheadb  = (__bf16*)wsalloc((size_t)4 * 2048 * 512 * 2);
    __bf16* hs      = (__bf16*)wsalloc((size_t)1024 * 8 * 8 * 512 * 2);   // [t][b][n][H]
    float*  q_ws    = (float*) wsalloc((size_t)8192 * 512 * 4);
    __bf16* ao      = (__bf16*)wsalloc((size_t)8192 * 512 * 2);
    float*  aoo     = (float*) wsalloc((size_t)8192 * 512 * 4);
    __bf16* mixed   = (__bf16*)wsalloc((size_t)8192 * 512 * 2);
    __bf16* stack   = (__bf16*)wsalloc((size_t)8192 * 8 * 512 * 2);       // [(b*T+t)*8+n][D]
    float*  proj_ws = (float*) wsalloc((size_t)8192 * 8 * 512 * 4);       // [t*8+b][n*512+d]
    // proj_ws is dead after stack_ln: reuse its 134MB for kk|vv (67MB each)
    __bf16* kkw = (__bf16*)proj_ws;
    __bf16* vvw = (__bf16*)((char*)proj_ws + (size_t)8192 * 8 * 512 * 2);

    // ---- weight conversion to bf16 ----
    wcat_kernel<<<65536, 256, 0, stream>>>(w_ih, w_hh, wcat);
    cvt_f32_bf16_kernel<<<8192, 256, 0, stream>>>(w_proj, wprojb, 8 * 512 * 512);
    cvt_f32_bf16_kernel<<<1024, 256, 0, stream>>>(wq, wqb, 512 * 512);
    cvt_f32_bf16_kernel<<<1024, 256, 0, stream>>>(wk, wkb, 512 * 512);
    cvt_f32_bf16_kernel<<<1024, 256, 0, stream>>>(wv, wvb, 512 * 512);
    cvt_f32_bf16_kernel<<<1024, 256, 0, stream>>>(wo, wob, 512 * 512);
    cvt_f32_bf16_kernel<<<16384, 256, 0, stream>>>(w_head, wheadb, 4 * 2048 * 512);

    // ---- embedding + LN ----
    embed_ln_kernel<<<8192, 256, 0, stream>>>(x, emb, ln_in_g, ln_in_b, xf_f32, xf_bt, xf_tb);

    // ---- LSTM scan (8 neurons -> 8 persistent workgroups) ----
    lstm_scan_kernel<<<NN, 512, 0, stream>>>(xf_tb, wcat, b_ih, b_hh, hs);

    // ---- per-neuron projection:  proj[r][n*512+d] = hs[r][n] @ w_proj[n]^T + b_proj[n]
    for (int n = 0; n < NN; ++n) {
        gemm_bf16_wmma_kernel<<<dim3(8, 512), 32, 0, stream>>>(
            hs + (size_t)n * 512, 4096,
            wprojb + (size_t)n * 512 * 512, 512,
            b_proj + n * 512,
            proj_ws + (size_t)n * 512, 4096, 512, 0);
    }
    stack_ln_kernel<<<65536, 128, 0, stream>>>(proj_ws, ln_n_g, ln_n_b, stack);

    // ---- attention projections ----
    gemm_bf16_wmma_kernel<<<dim3(8, 512), 32, 0, stream>>>(
        xf_bt, 512, wqb, 512, bq, q_ws, 512, 512, 0);
    gemm_bf16_wmma_kernel<<<dim3(8, 4096), 32, 0, stream>>>(
        stack, 512, wkb, 512, bk, kkw, 512, 512, 1);
    gemm_bf16_wmma_kernel<<<dim3(8, 4096), 32, 0, stream>>>(
        stack, 512, wvb, 512, bv, vvw, 512, 512, 1);
    attn_kernel<<<8192, 256, 0, stream>>>(q_ws, kkw, vvw, ao);
    gemm_bf16_wmma_kernel<<<dim3(8, 512), 32, 0, stream>>>(
        ao, 512, wob, 512, bo, aoo, 512, 512, 0);

    // ---- mix + LN ----
    mixed_ln_kernel<<<8192, 256, 0, stream>>>(aoo, xf_f32, ln_mix_g, ln_mix_b, mixed);

    // ---- head GEMM -> d_out [B,K,T,V] f32 (mode 2 scatter) ----
    gemm_bf16_wmma_kernel<<<dim3(128, 512), 32, 0, stream>>>(
        mixed, 512, wheadb, 512, b_head, d_out, 0, 512, 2);
}
